// dMaSIFConv_seg_29858612642361
// MI455X (gfx1250) — matrix operand order repaired and checked by reading, not compiled, not run on hardware
//
#include <hip/hip_runtime.h>

#define NPTS 2048
#define HCH  128

typedef __attribute__((ext_vector_type(16))) __bf16 v16bf;
typedef __attribute__((ext_vector_type(8)))  __bf16 v8bf;
typedef __attribute__((ext_vector_type(8)))  float  v8f;

__device__ __forceinline__ float lrelu(float x) { return x > 0.f ? x : 0.2f * x; }

// ============================================================================
// out = lrelu( lrelu(x @ W1^T + b1) @ W2^T + b2 ), one point per 128-thr block
// ============================================================================
__global__ __launch_bounds__(128) void mlp2_kernel(
    const float* __restrict__ xin, int din,
    const float* __restrict__ W1, const float* __restrict__ b1,
    const float* __restrict__ W2, const float* __restrict__ b2,
    float* __restrict__ out)
{
  const int i = blockIdx.x, h = threadIdx.x;
  __shared__ float sx[HCH], sf[HCH];
  if (h < din) sx[h] = xin[i * din + h];
  __syncthreads();
  float a = b1[h];
  const float* w = W1 + h * din;
  for (int k = 0; k < din; ++k) a += sx[k] * w[k];
  sf[h] = lrelu(a);
  __syncthreads();
  float o = b2[h];
  const float* w2 = W2 + h * HCH;
#pragma unroll 8
  for (int k = 0; k < HCH; ++k) o += sf[k] * w2[k];
  out[i * HCH + h] = lrelu(o);
}

// ============================================================================
// GroupNorm stats: group g = channels [32g,32g+32) over all N points.
// stats[2g] = mean, stats[2g+1] = rstd
// ============================================================================
__global__ __launch_bounds__(256) void gn_stats_kernel(
    const float* __restrict__ x, float* __restrict__ stats)
{
  const int g = blockIdx.x, t = threadIdx.x;
  float s = 0.f, ss = 0.f;
  for (int idx = t; idx < 32 * NPTS; idx += 256) {
    const int c = (g << 5) + (idx & 31);
    const int n = idx >> 5;
    const float v = x[n * HCH + c];
    s += v; ss += v * v;
  }
  __shared__ float sh0[256], sh1[256];
  sh0[t] = s; sh1[t] = ss;
  __syncthreads();
  for (int off = 128; off > 0; off >>= 1) {
    if (t < off) { sh0[t] += sh0[t + off]; sh1[t] += sh1[t + off]; }
    __syncthreads();
  }
  if (t == 0) {
    const float inv = 1.f / (32.f * NPTS);
    const float mu  = sh0[0] * inv;
    const float var = sh1[0] * inv - mu * mu;
    stats[g * 2 + 0] = mu;
    stats[g * 2 + 1] = rsqrtf(var + 1e-5f);
  }
}

// ============================================================================
// fT[h][j] = bf16( GN(f)[j,h] * gamma[h] + beta[h] )   (transposed, bf16)
// ============================================================================
__global__ __launch_bounds__(256) void make_fT_kernel(
    const float* __restrict__ f, const float* __restrict__ stats,
    const float* __restrict__ gamma, const float* __restrict__ beta,
    __bf16* __restrict__ fT)
{
  const int idx = blockIdx.x * 256 + threadIdx.x;   // idx = h*NPTS + j
  const int h = idx >> 11, j = idx & (NPTS - 1);
  const int g = h >> 5;
  const float v = (f[j * HCH + h] - stats[g * 2]) * stats[g * 2 + 1] * gamma[h] + beta[h];
  fT[idx] = (__bf16)v;
}

// ============================================================================
// Heavy kernel v2: 128 threads = 4 waves; 8 points per block (2 per wave).
// Per 32-j chunk: the block stages the 32x128 bf16 fT tile into LDS once
// (8x reuse vs per-wave global streaming), each wave computes pair geometry
// for its two points, builds two bf16 A-fragments (8 relu-cut rows, row8 =
// window, rows 9..15 = 0), and issues 16 v_wmma_f32_16x16x32_bf16 per chunk
// (2 independent accumulator sets sharing each B fragment).
// Epilogue: conv[i,h] = sum_c A2[h,c]*T[c,h] + B2[h]*T[8,h].
// ============================================================================
__global__ __launch_bounds__(128) void dmasif_conv_wmma_kernel(
    const float* __restrict__ pts, const float* __restrict__ nuv,
    const __bf16* __restrict__ fT,
    const float* __restrict__ A1, const float* __restrict__ B1,
    const float* __restrict__ A2, const float* __restrict__ B2,
    float* __restrict__ conv_out)
{
  const int tid  = threadIdx.x;
  const int lane = tid & 31;
  const int wave = tid >> 5;
  const int i0   = blockIdx.x * 8 + wave * 2;   // this wave's two points
  const int i1   = i0 + 1;
  const int r    = lane & 15;   // A-matrix row owned by this lane
  const int half = lane >> 4;   // 0: lanes 0-15, 1: lanes 16-31

  __shared__ __bf16 btile[HCH][32];     // staged fT chunk  [h][j-j0]  (8 KB)
  __shared__ float4 pair[4][2][32];     // per wave, per point: {w,X0,X1,X2}

  const float ps = 0.0785674201318385f; // 1 / (sqrt(2) * RADIUS)
  const float pax = pts[i0*3+0]*ps, pay = pts[i0*3+1]*ps, paz = pts[i0*3+2]*ps;
  const float pbx = pts[i1*3+0]*ps, pby = pts[i1*3+1]*ps, pbz = pts[i1*3+2]*ps;
  const float an0 = nuv[i0*9+0], an1 = nuv[i0*9+1], an2 = nuv[i0*9+2];
  const float au0 = nuv[i0*9+3], au1 = nuv[i0*9+4], au2 = nuv[i0*9+5];
  const float av0 = nuv[i0*9+6], av1 = nuv[i0*9+7], av2 = nuv[i0*9+8];
  const float bn0 = nuv[i1*9+0], bn1 = nuv[i1*9+1], bn2 = nuv[i1*9+2];
  const float bu0 = nuv[i1*9+3], bu1 = nuv[i1*9+4], bu2 = nuv[i1*9+5];
  const float bv0 = nuv[i1*9+6], bv1 = nuv[i1*9+7], bv2 = nuv[i1*9+8];

  float a1x = 0.f, a1y = 0.f, a1z = 0.f, b1r = 0.f;
  if (r < 8) { a1x = A1[r*3+0]; a1y = A1[r*3+1]; a1z = A1[r*3+2]; b1r = B1[r]; }

  // 16-bit A fragment K interleave: lanes 0-15 pack K{0..7,16..23},
  // lanes 16-31 pack K{8..15,24..31}. 16-bit B: same interleave as A.
  const int jsel0 = half ? 8 : 0;
  const int jsel1 = jsel0 + 16;
  const int bsel  = half ? 8 : 0;

  v8f acc0[8], acc1[8];
  const v8f vzero = {};
#pragma unroll
  for (int t = 0; t < 8; ++t) { acc0[t] = vzero; acc1[t] = vzero; }

  for (int j0 = 0; j0 < NPTS; j0 += 32) {
    __syncthreads();   // all waves done consuming previous tile

    // ---- stage fT[h][j0..j0+31] into LDS: thread tid owns row h = tid ----
    {
      const __bf16* src = fT + (size_t)tid * NPTS + j0;
      const float4 q0 = *reinterpret_cast<const float4*>(src);
      const float4 q1 = *reinterpret_cast<const float4*>(src + 8);
      const float4 q2 = *reinterpret_cast<const float4*>(src + 16);
      const float4 q3 = *reinterpret_cast<const float4*>(src + 24);
      float4* dst = reinterpret_cast<float4*>(&btile[tid][0]);
      dst[0] = q0; dst[1] = q1; dst[2] = q2; dst[3] = q3;
    }
    if (j0 + 32 < NPTS)
      __builtin_prefetch(fT + (size_t)tid * NPTS + j0 + 32, 0, 0);

    // ---- pair geometry for this wave's two points, j = j0 + lane ----
    {
      const int j = j0 + lane;
      const float qx = pts[j*3+0]*ps, qy = pts[j*3+1]*ps, qz = pts[j*3+2]*ps;
      const float mx = nuv[j*9+0],  my = nuv[j*9+1],  mz = nuv[j*9+2];

      float dx = qx - pax, dy = qy - pay, dz = qz - paz;
      float nd = 2.f - (an0*mx + an1*my + an2*mz);
      float d2 = (dx*dx + dy*dy + dz*dz) * nd * nd;
      float4 pj;
      pj.x = __expf(-d2);
      pj.y = an0*dx + an1*dy + an2*dz;
      pj.z = au0*dx + au1*dy + au2*dz;
      pj.w = av0*dx + av1*dy + av2*dz;
      pair[wave][0][lane] = pj;

      dx = qx - pbx; dy = qy - pby; dz = qz - pbz;
      nd = 2.f - (bn0*mx + bn1*my + bn2*mz);
      d2 = (dx*dx + dy*dy + dz*dz) * nd * nd;
      pj.x = __expf(-d2);
      pj.y = bn0*dx + bn1*dy + bn2*dz;
      pj.z = bu0*dx + bu1*dy + bu2*dz;
      pj.w = bv0*dx + bv1*dy + bv2*dz;
      pair[wave][1][lane] = pj;
    }
    __syncthreads();   // tile + pair visible to all consumers

    // ---- build the two bf16 A fragments ----
    v16bf afrag0, afrag1;
#pragma unroll
    for (int k = 0; k < 8; ++k) {
      const float4 p00 = pair[wave][0][jsel0 + k];
      const float4 p01 = pair[wave][0][jsel1 + k];
      const float4 p10 = pair[wave][1][jsel0 + k];
      const float4 p11 = pair[wave][1][jsel1 + k];
      float m00, m01, m10, m11;
      if (r < 8) {
        m00 = p00.x * fmaxf(a1x*p00.y + a1y*p00.z + a1z*p00.w + b1r, 0.f);
        m01 = p01.x * fmaxf(a1x*p01.y + a1y*p01.z + a1z*p01.w + b1r, 0.f);
        m10 = p10.x * fmaxf(a1x*p10.y + a1y*p10.z + a1z*p10.w + b1r, 0.f);
        m11 = p11.x * fmaxf(a1x*p11.y + a1y*p11.z + a1z*p11.w + b1r, 0.f);
      } else if (r == 8) { m00 = p00.x; m01 = p01.x; m10 = p10.x; m11 = p11.x; }
      else               { m00 = m01 = m10 = m11 = 0.f; }
      afrag0[k]     = (__bf16)m00;
      afrag0[k + 8] = (__bf16)m01;
      afrag1[k]     = (__bf16)m10;
      afrag1[k + 8] = (__bf16)m11;
    }

    // ---- 8 B fragments from LDS, 16 WMMAs (2 per fragment) ----
#pragma unroll
    for (int ht = 0; ht < 8; ++ht) {
      const __bf16* row = &btile[ht * 16 + r][bsel];
      const v8bf blo = *reinterpret_cast<const v8bf*>(row);
      const v8bf bhi = *reinterpret_cast<const v8bf*>(row + 16);
      v16bf bfrag;
#pragma unroll
      for (int k = 0; k < 8; ++k) { bfrag[k] = blo[k]; bfrag[k + 8] = bhi[k]; }
      acc0[ht] = __builtin_amdgcn_wmma_f32_16x16x32_bf16(
          false, afrag0, false, bfrag, (short)0, acc0[ht], false, false);
      acc1[ht] = __builtin_amdgcn_wmma_f32_16x16x32_bf16(
          false, afrag1, false, bfrag, (short)0, acc1[ht], false, false);
    }
  }

  // ---- epilogue: lanes 0-15 hold T[c=0..7, h]; lane l+16 VGPR0 holds T[8,h] ----
#pragma unroll
  for (int ht = 0; ht < 8; ++ht) {
    const float t8a = __shfl(acc0[ht][0], r + 16, 32);
    const float t8b = __shfl(acc1[ht][0], r + 16, 32);
    if (lane < 16) {
      const int h = ht * 16 + r;
      float ca = B2[h] * t8a;
      float cb = B2[h] * t8b;
#pragma unroll
      for (int cc = 0; cc < 8; ++cc) {
        const float a2 = A2[h * 8 + cc];
        ca += a2 * acc0[ht][cc];
        cb += a2 * acc1[ht][cc];
      }
      conv_out[i0 * HCH + h] = ca;
      conv_out[i1 * HCH + h] = cb;
    }
  }
}

// ============================================================================
// tail: gn = GN(g_pre)*gamma+beta ; xi = relu(gn@W1^T+b1)@W2^T+b2 ;
//       xout = xin@Wtr^T+btr + xi
// ============================================================================
__global__ __launch_bounds__(128) void fuse_tail_kernel(
    const float* __restrict__ g_pre, const float* __restrict__ stats,
    const float* __restrict__ gamma, const float* __restrict__ beta,
    const float* __restrict__ W1, const float* __restrict__ b1,
    const float* __restrict__ W2, const float* __restrict__ b2,
    const float* __restrict__ xin, int din,
    const float* __restrict__ Wtr, const float* __restrict__ btr,
    float* __restrict__ xout)
{
  const int i = blockIdx.x, h = threadIdx.x;
  __shared__ float sg[HCH], st[HCH], sx[HCH];
  const int gg = h >> 5;
  sg[h] = (g_pre[i * HCH + h] - stats[gg * 2]) * stats[gg * 2 + 1] * gamma[h] + beta[h];
  if (h < din) sx[h] = xin[i * din + h];
  __syncthreads();
  float t = b1[h];
  const float* w1 = W1 + h * HCH;
#pragma unroll 8
  for (int k = 0; k < HCH; ++k) t += sg[k] * w1[k];
  st[h] = fmaxf(t, 0.f);
  __syncthreads();
  float xi = b2[h];
  const float* w2 = W2 + h * HCH;
#pragma unroll 8
  for (int k = 0; k < HCH; ++k) xi += st[k] * w2[k];
  float xt = btr[h];
  const float* wt = Wtr + h * din;
  for (int k = 0; k < din; ++k) xt += sx[k] * wt[k];
  xout[i * HCH + h] = xt + xi;
}

// ============================================================================
extern "C" void kernel_launch(void* const* d_in, const int* in_sizes, int n_in,
                              void* d_out, int out_size, void* d_ws, size_t ws_size,
                              hipStream_t stream)
{
  (void)out_size; (void)ws_size;

  // ---- locate top-level inputs (handles insertion-order or pytree-sorted) ----
  const float *features, *points, *nuv;
  int pbase;
  if (in_sizes[1] == NPTS * 3) {           // insertion: features, points, nuv, params...
    features = (const float*)d_in[0];
    points   = (const float*)d_in[1];
    nuv      = (const float*)d_in[2];
    pbase = 3;
  } else {                                 // sorted: features, nuv, params..., points
    features = (const float*)d_in[0];
    nuv      = (const float*)d_in[1];
    pbase = 2;
    points   = (const float*)d_in[n_in - 1];
  }

  // ---- leaf ordering inside each conv dict ----
  // fields: A1,A2,B1,B2,W_in1,W_in2,W_o1,W_o2,b_in1,b_in2,b_o1,b_o2,be_in,be_out,g_in,g_out
  const bool leavesSorted = (in_sizes[pbase] == 24);  // A1 (8x3) first <=> sorted keys
  static const int MAP_SORT[16] = {0,1,2,3,4,5,6,7,8,9,10,11,12,13,14,15};
  static const int MAP_INS [16] = {6,8,7,9,0,2,10,12,1,3,11,13,5,15,4,14};
  const int* cm = leavesSorted ? MAP_SORT : MAP_INS;
  // lin_mlp fields {W1,b1,W2,b2}
  static const int MLP_SORT[4] = {0,2,1,3};
  static const int MLP_INS [4] = {0,1,2,3};
  const int* mm = leavesSorted ? MLP_SORT : MLP_INS;

  // ---- workspace layout (floats) ----
  float*  ws       = (float*)d_ws;
  float*  f_pre    = ws;                  // 262144
  float*  conv_out = ws + 262144;         // 262144
  float*  g_pre    = ws + 524288;         // 262144
  float*  xA       = ws + 786432;         // 262144
  float*  xB       = ws + 1048576;        // 262144
  float*  stats    = ws + 1310720;        // 16
  __bf16* fT       = (__bf16*)(ws + 1310736); // 262144 bf16 (16B aligned)

  const float* xin = features;
  int din = 16;
  for (int L = 0; L < 3; ++L) {
    const int cb = pbase + 16 * L;        // conv dict leaves
    const int mb = pbase + 48 + 4 * L;    // lin_mlp leaves
    const int tb = pbase + 60 + 2 * L;    // lin_tr leaves
    const float* A1    = (const float*)d_in[cb + cm[0]];
    const float* A2    = (const float*)d_in[cb + cm[1]];
    const float* B1    = (const float*)d_in[cb + cm[2]];
    const float* B2    = (const float*)d_in[cb + cm[3]];
    const float* W_in1 = (const float*)d_in[cb + cm[4]];
    const float* W_in2 = (const float*)d_in[cb + cm[5]];
    const float* W_o1  = (const float*)d_in[cb + cm[6]];
    const float* W_o2  = (const float*)d_in[cb + cm[7]];
    const float* b_in1 = (const float*)d_in[cb + cm[8]];
    const float* b_in2 = (const float*)d_in[cb + cm[9]];
    const float* b_o1  = (const float*)d_in[cb + cm[10]];
    const float* b_o2  = (const float*)d_in[cb + cm[11]];
    const float* be_in = (const float*)d_in[cb + cm[12]];
    const float* be_out= (const float*)d_in[cb + cm[13]];
    const float* g_in  = (const float*)d_in[cb + cm[14]];
    const float* g_out = (const float*)d_in[cb + cm[15]];
    const float* W1m   = (const float*)d_in[mb + mm[0]];
    const float* b1m   = (const float*)d_in[mb + mm[1]];
    const float* W2m   = (const float*)d_in[mb + mm[2]];
    const float* b2m   = (const float*)d_in[mb + mm[3]];
    const float* Wtr   = (const float*)d_in[tb + 0];
    const float* btr   = (const float*)d_in[tb + 1];

    float* xout = (L == 0) ? xA : (L == 1) ? xB : (float*)d_out;

    mlp2_kernel<<<NPTS, 128, 0, stream>>>(xin, din, W_in1, b_in1, W_in2, b_in2, f_pre);
    gn_stats_kernel<<<4, 256, 0, stream>>>(f_pre, stats);
    make_fT_kernel<<<(NPTS * HCH) / 256, 256, 0, stream>>>(f_pre, stats, g_in, be_in, fT);
    dmasif_conv_wmma_kernel<<<NPTS / 8, 128, 0, stream>>>(points, nuv, fT, A1, B1, A2, B2, conv_out);
    mlp2_kernel<<<NPTS, 128, 0, stream>>>(conv_out, HCH, W_o1, b_o1, W_o2, b_o2, g_pre);
    gn_stats_kernel<<<4, 256, 0, stream>>>(g_pre, stats + 8);
    fuse_tail_kernel<<<NPTS, 128, 0, stream>>>(g_pre, stats + 8, g_out, be_out,
                                               W1m, b1m, W2m, b2m, xin, din, Wtr, btr, xout);
    xin = xout;
    din = HCH;
  }
}
